// QuantumLayer_816043786806
// MI455X (gfx1250) — matrix operand order repaired and checked by reading, not compiled, not run on hardware
//
#include <hip/hip_runtime.h>
#include <hip/hip_bf16.h>

#define NQ      12
#define DIMS    4096       // 1 << NQ
#define NL      4
#define BATCHN  256
#define THREADS 256

typedef _Float16 half8  __attribute__((ext_vector_type(8)));
typedef _Float16 half16 __attribute__((ext_vector_type(16)));
typedef float    float8 __attribute__((ext_vector_type(8)));

// ---------------------------------------------------------------------------
// Kernel 1: per-batch statevector simulation, LDS-resident.
//   - fused U = RZ*RY*RX per (layer,qubit), precomputed cooperatively
//   - CZ ring as one parity sign pass per layer
//   - writes |amp|^2 as f16 probs to workspace: probs[b*4096 + s]
// ---------------------------------------------------------------------------
__global__ __launch_bounds__(THREADS)
void qsim_kernel(const float* __restrict__ x,      // (BATCH, NQ)
                 const float* __restrict__ w,      // (NL, NQ, 3)
                 _Float16* __restrict__ probs)     // (BATCH, DIMS) f16
{
    __shared__ float sr[DIMS];
    __shared__ float si[DIMS];
    __shared__ float uu[NL * NQ * 8];   // fused gate: 4 complex entries
    __shared__ float enc[NQ * 2];       // (c, s) for encoding RX per qubit

    const int b = blockIdx.x;
    const int t = threadIdx.x;

    // ---- cooperative precompute of fused layer gates -----------------------
    if (t < NL * NQ) {
        const int l = t / NQ, q = t % NQ;
        const float ha = 0.5f * w[(l * NQ + q) * 3 + 0];
        const float hb = 0.5f * w[(l * NQ + q) * 3 + 1];
        const float hg = 0.5f * w[(l * NQ + q) * 3 + 2];
        float sa, ca, sb, cb, sg, cg;
        sincosf(ha, &sa, &ca);
        sincosf(hb, &sb, &cb);
        sincosf(hg, &sg, &cg);
        // M = RY * RX
        const float m00r =  cb * ca, m00i =  sb * sa;
        const float m01r = -sb * ca, m01i = -cb * sa;
        const float m10r =  sb * ca, m10i = -cb * sa;
        const float m11r =  cb * ca, m11i = -sb * sa;
        // U = diag(cg - i sg, cg + i sg) * M
        float* U = &uu[t * 8];
        U[0] = cg * m00r + sg * m00i;  U[1] = cg * m00i - sg * m00r;   // U00
        U[2] = cg * m01r + sg * m01i;  U[3] = cg * m01i - sg * m01r;   // U01
        U[4] = cg * m10r - sg * m10i;  U[5] = cg * m10i + sg * m10r;   // U10
        U[6] = cg * m11r - sg * m11i;  U[7] = cg * m11i + sg * m11r;   // U11
    }
    if (t >= 64 && t < 64 + NQ) {
        const int q = t - 64;
        float s, c;
        sincosf(0.5f * x[b * NQ + q], &s, &c);
        enc[q * 2 + 0] = c;
        enc[q * 2 + 1] = s;
    }
    // ---- init |0...0> ------------------------------------------------------
    for (int i = t; i < DIMS; i += THREADS) { sr[i] = 0.0f; si[i] = 0.0f; }
    if (t == 0) sr[0] = 1.0f;
    __syncthreads();

    // ---- butterfly: apply 2x2 complex U on wire q --------------------------
    auto butterfly = [&](int q,
                         float u00r, float u00i, float u01r, float u01i,
                         float u10r, float u10i, float u11r, float u11i) {
        const int shift   = NQ - 1 - q;          // wire 0 = MSB of state index
        const int lowmask = (1 << shift) - 1;
        for (int p = t; p < DIMS / 2; p += THREADS) {
            const int i0 = ((p & ~lowmask) << 1) | (p & lowmask);
            const int i1 = i0 | (1 << shift);
            const float r0 = sr[i0], m0 = si[i0];
            const float r1 = sr[i1], m1 = si[i1];
            sr[i0] = u00r * r0 - u00i * m0 + u01r * r1 - u01i * m1;
            si[i0] = u00r * m0 + u00i * r0 + u01r * m1 + u01i * r1;
            sr[i1] = u10r * r0 - u10i * m0 + u11r * r1 - u11i * m1;
            si[i1] = u10r * m0 + u10i * r0 + u11r * m1 + u11i * r1;
        }
        __syncthreads();
    };

    // ---- angle encoding: RX(x[b][q]) on wire q -----------------------------
    for (int q = 0; q < NQ; ++q) {
        const float c = enc[q * 2 + 0], s = enc[q * 2 + 1];
        butterfly(q, c, 0.0f, 0.0f, -s, 0.0f, -s, c, 0.0f);
    }

    // ---- layers: fused RX/RY/RZ, then CZ ring ------------------------------
    for (int l = 0; l < NL; ++l) {
        for (int q = 0; q < NQ; ++q) {
            const float* U = &uu[(l * NQ + q) * 8];
            butterfly(q, U[0], U[1], U[2], U[3], U[4], U[5], U[6], U[7]);
        }
        // CZ(i, (i+1)%12) ring: sign = (-1)^(# adjacent bit pairs set)
        for (int i = t; i < DIMS; i += THREADS) {
            const int adj  = (i & (i >> 1)) & 0x7FF;   // pairs q=0..10
            const int wrap = (i & (i >> 11)) & 1;      // pair (11, 0)
            if ((__popc(adj) + wrap) & 1) { sr[i] = -sr[i]; si[i] = -si[i]; }
        }
        __syncthreads();
    }

    // ---- probs -> f16 workspace -------------------------------------------
    _Float16* outp = probs + (size_t)b * DIMS;
    for (int i = t; i < DIMS; i += THREADS) {
        outp[i] = (_Float16)(sr[i] * sr[i] + si[i] * si[i]);
    }
}

// ---------------------------------------------------------------------------
// Kernel 2: <Z_n> = probs @ Z_TABLE via v_wmma_f32_16x16x32_f16.
//   One wave32 per 16-row (batch) tile; N padded 12 -> 16 (cols 12..15 = 0).
//   Z-table structure: within a 32-wide K chunk, columns 7..11 have a
//   kb-invariant sign pattern and columns 0..6 are a uniform sign flip of an
//   all-(+1) pattern by bit (11-n) of kb. So the B fragment each iteration is
//   base_fragment ^ broadcast_sign -- 1 cndmask + 8 XORs per WMMA.
// ---------------------------------------------------------------------------
__global__ __launch_bounds__(32)
void zexp_wmma_kernel(const _Float16* __restrict__ P,   // (BATCH, DIMS) f16
                      float* __restrict__ out)          // (BATCH, NQ) f32
{
    const int lane   = threadIdx.x;
    const int mt     = blockIdx.x;        // 16 tiles of 16 batch rows
    const int col    = lane & 15;         // N index (output qubit)
    const int hiHalf = lane >> 4;         // lane group 0/1
    const int row    = mt * 16 + col;     // A-matrix row for this lane
    const int koffA  = hiHalf ? 8 : 0;    // A: halves sub-offset in 32-chunk
    const int kbB    = hiHalf ? 16 : 0;   // B: 2x K sub-offset in 32-chunk (used as kbB>>1)

    // Z-table generation constants (uniform per lane, valid for col < 12)
    const int      sh    = (col < NQ) ? (NQ - 1 - col) : 0;
    const unsigned base  = (col < NQ) ? 0x3C00u : 0u;   // f16 +1.0
    const unsigned smask = (col < NQ) ? 0x8000u : 0u;   // f16 sign bit

    // Base B fragment at kb = 0 (low-5-bit pattern; kb-invariant part).
    union { half16 h; unsigned u[8]; } F;
#pragma unroll
    for (int v = 0; v < 8; ++v) {
        const unsigned k0  = (unsigned)((kbB >> 1) + 2 * v + (v >= 4 ? 8 : 0));
        const unsigned k1  = k0 + 1;
        const unsigned l16 = base | (((k0 >> sh) & 1u) ? smask : 0u);
        const unsigned h16 = base | (((k1 >> sh) & 1u) ? smask : 0u);
        F.u[v] = l16 | (h16 << 16);
    }

    float8 acc = {0.f, 0.f, 0.f, 0.f, 0.f, 0.f, 0.f, 0.f};
    const _Float16* rowp = P + (size_t)row * DIMS;

    for (int kb = 0; kb < DIMS; kb += 32) {
        // A fragment: 16-bit 16x32 layout -> two contiguous 16B half8 chunks
        const half8 lo = *(const half8*)(rowp + kb + koffA);
        const half8 hi = *(const half8*)(rowp + kb + 16 + koffA);
        half16 a;
#pragma unroll
        for (int i = 0; i < 8; ++i) { a[i] = lo[i]; a[8 + i] = hi[i]; }

        // B fragment = base pattern ^ uniform sign of bit sh of kb.
        // (For sh < 5 and for padded cols, bit sh of kb is always 0.)
        const unsigned sx = (((unsigned)kb >> sh) & 1u) ? 0x80008000u : 0u;
        union { half16 h; unsigned u[8]; } bf;
#pragma unroll
        for (int v = 0; v < 8; ++v) bf.u[v] = F.u[v] ^ sx;

        acc = __builtin_amdgcn_wmma_f32_16x16x32_f16(
                  /*neg_a=*/false, a, /*neg_b=*/false, bf.h,
                  /*c_mod=*/(short)0, acc, /*reuse_a=*/false, /*reuse_b=*/false);
    }

    // D layout: VGPR r holds M = r (lanes 0-15) / M = 8+r (lanes 16-31), N=col
    if (col < NQ) {
#pragma unroll
        for (int r = 0; r < 8; ++r) {
            const int m = mt * 16 + r + (hiHalf ? 8 : 0);
            out[m * NQ + col] = acc[r];
        }
    }
}

// ---------------------------------------------------------------------------
extern "C" void kernel_launch(void* const* d_in, const int* in_sizes, int n_in,
                              void* d_out, int out_size, void* d_ws, size_t ws_size,
                              hipStream_t stream)
{
    const float* x = (const float*)d_in[0];   // (256, 12) f32
    const float* w = (const float*)d_in[1];   // (4, 12, 3) f32
    _Float16* probs = (_Float16*)d_ws;        // needs BATCH*DIMS*2 = 2 MB

    qsim_kernel<<<BATCHN, THREADS, 0, stream>>>(x, w, probs);
    zexp_wmma_kernel<<<BATCHN / 16, 32, 0, stream>>>(probs, (float*)d_out);
}
